// ALE_8899172237562
// MI455X (gfx1250) — matrix operand order repaired and checked by reading, not compiled, not run on hardware
//
#include <hip/hip_runtime.h>

// ---- CDNA5 WMMA vector types (wave32) ----
typedef __attribute__((ext_vector_type(16))) __bf16 v16bf;
typedef __attribute__((ext_vector_type(8)))  float  v8f;
typedef __attribute__((ext_vector_type(4)))  __bf16 v4bf;

union BFrag { v16bf v; uint4 q[2]; };

// ---------------- fp32 -> bf16 conversion kernels ----------------

// x [B, IMG] contiguous convert (vectorized 4-wide)
__global__ void ALE_cvt_x(const float* __restrict__ x, __bf16* __restrict__ xb, int n4) {
  int i = blockIdx.x * blockDim.x + threadIdx.x;
  if (i >= n4) return;
  float4 f = ((const float4*)x)[i];
  v4bf r;
  r[0] = (__bf16)f.x; r[1] = (__bf16)f.y; r[2] = (__bf16)f.z; r[3] = (__bf16)f.w;
  ((v4bf*)xb)[i] = r;
}

// W [K=2048, N=300] -> Wt_bf16 [Npad=320, K=2048], transposed + zero-padded rows
__global__ void ALE_cvt_Wt(const float* __restrict__ W, __bf16* __restrict__ Wt,
                           int K, int Npad, int N) {
  int i = blockIdx.x * blockDim.x + threadIdx.x;
  if (i >= Npad * K) return;
  int n = i / K;
  int k = i - n * K;
  Wt[i] = (n < N) ? (__bf16)W[(size_t)k * N + n] : (__bf16)0.0f;
}

// E [C, K=300] -> Eb [C, Kpad=320], zero-padded K tail
__global__ void ALE_cvt_E(const float* __restrict__ E, __bf16* __restrict__ Eb,
                          int C, int Kpad, int K) {
  size_t i = (size_t)blockIdx.x * blockDim.x + threadIdx.x;
  if (i >= (size_t)C * Kpad) return;
  int    k = (int)(i % Kpad);
  size_t c = i / Kpad;
  Eb[i] = (k < K) ? (__bf16)E[c * (size_t)K + k] : (__bf16)0.0f;
}

// ---------------- wave-level bf16 WMMA GEMM, 32x80 tile per wave ----------------
// C[M,N] = A[M,K] * B[K,N], with B supplied "column-major" as Bt[N, ldb]
// (row n of Bt holds column n of B => contiguous K per lane).
// Each wave computes a 32x80 output tile: 2 A fragments (M) x 5 B fragments (N)
// = 10 accumulators. All 14 b128 fragment loads of a k-step are issued as one
// clause BEFORE the 10-WMMA burst, so later B loads overlap earlier WMMAs
// (staggered s_wait_loadcnt instead of wait-0 per WMMA pair).
template <typename OutT>
__global__ __launch_bounds__(256) void ALE_wmma_gemm(
    const __bf16* __restrict__ A, const __bf16* __restrict__ Bt,
    OutT* __restrict__ C, int Mtiles /* of 32 rows */, int Ngroups /* of 80 cols */,
    int K, int lda, int ldb, int ldc) {
  const int lane = threadIdx.x & 31;
  const int wid  = blockIdx.x * (blockDim.x >> 5) + (threadIdx.x >> 5);
  if (wid >= Mtiles * Ngroups) return;
  const int mt = wid % Mtiles;
  const int ng = wid / Mtiles;
  const int m0 = mt << 5;      // 32 rows per wave tile
  const int n0 = ng * 80;      // 80 cols per wave tile

  const int l15 = lane & 15;
  const int hi  = lane >> 4;   // half-wave select

  // A fragment sources: row m = m0 + l15 (+16); lanes 0-15 K+0, lanes 16-31 K+8
  const __bf16* aRow0 = A + (size_t)(m0 + l15) * lda + (hi << 3);
  const __bf16* aRow1 = aRow0 + (size_t)16 * lda;
  // B fragment sources: column n = n0 + 16j + l15; lanes 0-15 K+0..7,16..23, 16-31 K+8..15,24..31
  const __bf16* bRow[5];
#pragma unroll
  for (int j = 0; j < 5; ++j)
    bRow[j] = Bt + (size_t)(n0 + (j << 4) + l15) * ldb + (hi << 3);

  v8f acc0[5] = {};   // rows m0 .. m0+15
  v8f acc1[5] = {};   // rows m0+16 .. m0+31

  for (int k0 = 0; k0 < K; k0 += 32) {
    // ---- load phase: 14 x global_load_b128, one clause, in consume order ----
    BFrag a0, a1, b[5];
    a0.q[0] = *(const uint4*)(aRow0 + k0);        // K = k0 + base .. +7
    a0.q[1] = *(const uint4*)(aRow0 + k0 + 16);   // K = k0 + 16 + base .. +23
    a1.q[0] = *(const uint4*)(aRow1 + k0);
    a1.q[1] = *(const uint4*)(aRow1 + k0 + 16);
#pragma unroll
    for (int j = 0; j < 5; ++j) {
      b[j].q[0] = *(const uint4*)(bRow[j] + k0);     // K = k0 + base .. +7
      b[j].q[1] = *(const uint4*)(bRow[j] + k0 + 16);
    }
    if (k0 + 32 < K) {
      __builtin_prefetch(aRow0 + k0 + 32, 0, 1);  // global_prefetch_b8
      __builtin_prefetch(aRow1 + k0 + 32, 0, 1);
    }
    // ---- compute phase: 10 x v_wmma_f32_16x16x32_bf16 ----
#pragma unroll
    for (int j = 0; j < 5; ++j) {
      acc0[j] = __builtin_amdgcn_wmma_f32_16x16x32_bf16(
          false, a0.v, false, b[j].v, (short)0, acc0[j], false, false);
      acc1[j] = __builtin_amdgcn_wmma_f32_16x16x32_bf16(
          false, a1.v, false, b[j].v, (short)0, acc1[j], false, false);
    }
  }

  // C/D layout: lane n = l15 (+16j); VGPR i -> row base + i + 8*hi
  const int mb0 = m0 + (hi << 3);
  const int mb1 = mb0 + 16;
#pragma unroll
  for (int j = 0; j < 5; ++j) {
    const int n = n0 + (j << 4) + l15;
#pragma unroll
    for (int i = 0; i < 8; ++i) {
      C[(size_t)(mb0 + i) * ldc + n] = (OutT)acc0[j][i];
      C[(size_t)(mb1 + i) * ldc + n] = (OutT)acc1[j][i];
    }
  }
}

extern "C" void kernel_launch(void* const* d_in, const int* in_sizes, int n_in,
                              void* d_out, int out_size, void* d_ws, size_t ws_size,
                              hipStream_t stream) {
  (void)in_sizes; (void)n_in; (void)out_size; (void)ws_size;
  const float* x = (const float*)d_in[0];           // [4096, 2048]
  const float* E = (const float*)d_in[1];           // [20000, 300]
  const float* W = (const float*)d_in[2];           // [2048, 300]
  float* out = (float*)d_out;                        // [4096, 20000]

  constexpr int B = 4096, IMG = 2048, WORD = 300, NCLS = 20000, WPAD = 320;

  // Workspace layout (all sizes already 256B-aligned)
  char* ws = (char*)d_ws;
  size_t off = 0;
  __bf16* xb = (__bf16*)(ws + off); off += (size_t)B * IMG * 2;      // 16.78 MB
  __bf16* Wt = (__bf16*)(ws + off); off += (size_t)WPAD * IMG * 2;   //  1.31 MB
  __bf16* Eb = (__bf16*)(ws + off); off += (size_t)NCLS * WPAD * 2;  // 12.80 MB
  __bf16* XW = (__bf16*)(ws + off);                                  //  2.62 MB

  // 1) Precision conversion + repack (HBM-bound, ~60 MB total traffic)
  {
    int n4 = B * IMG / 4;
    ALE_cvt_x<<<(n4 + 255) / 256, 256, 0, stream>>>(x, xb, n4);
    int nw = WPAD * IMG;
    ALE_cvt_Wt<<<(nw + 255) / 256, 256, 0, stream>>>(W, Wt, IMG, WPAD, WORD);
    long ne = (long)NCLS * WPAD;
    ALE_cvt_E<<<(int)((ne + 255) / 256), 256, 0, stream>>>(E, Eb, NCLS, WPAD, WORD);
  }

  // 2) GEMM1: XW[B, 320] = xb[B, 2048] @ W   (padded cols 300..319 come out zero)
  //    Mtiles = 4096/32 = 128, Ngroups = 320/80 = 4 -> 512 waves -> 64 blocks
  ALE_wmma_gemm<__bf16><<<(B / 32) * (WPAD / 80) / 8, 256, 0, stream>>>(
      xb, Wt, XW, B / 32, WPAD / 80, IMG, /*lda=*/IMG, /*ldb=*/IMG, /*ldc=*/WPAD);

  // 3) GEMM2: out[B, 20000] = XW[B, 320] @ Eb^T (K padded to 320, zeros inert)
  //    Mtiles = 128, Ngroups = 250 -> 32000 waves -> 4000 blocks
  ALE_wmma_gemm<float><<<(B / 32) * (NCLS / 80) / 8, 256, 0, stream>>>(
      XW, Eb, out, B / 32, NCLS / 80, WPAD, /*lda=*/WPAD, /*ldb=*/WPAD, /*ldc=*/NCLS);
}
